// APPNP_SR_52149492908282
// MI455X (gfx1250) — compile-verified, matrix operand order checked by reading, not run on hardware
//
#include <hip/hip_runtime.h>
#include <hip/hip_bf16.h>
#include <math.h>

typedef __attribute__((ext_vector_type(16))) __bf16 v16bf;
typedef __attribute__((ext_vector_type(8)))  __bf16 v8bf;
typedef __attribute__((ext_vector_type(8)))  float  v8f;

#define N_NODES 100000
#define F_IN    500
#define H_DIM   64
#define C_DIM   40
#define N_EDGES 3200000
#define K_STEPS 10
#define ALPHA   0.1f

// ---------------------------------------------------------------------------
// GEMM1: h1[N x 64] = relu(x[N x 500] @ W1[500 x 64] + b1), bf16 output.
// One wave per 16x16 output tile; 8 waves/block -> 2 row-tiles x 4 col-tiles.
// 15 unguarded k-steps (k<480) + one branchless peeled tail (k=480..499).
// ---------------------------------------------------------------------------
__global__ __launch_bounds__(256) void appnp_gemm1(const float* __restrict__ x,
                                                   const float* __restrict__ W1,
                                                   const float* __restrict__ b1,
                                                   __bf16* __restrict__ h1) {
    const int lane = threadIdx.x & 31;
    const int wid  = threadIdx.x >> 5;
    const int rowTile = blockIdx.x * 2 + (wid >> 2);   // 0..6249
    const int colTile = wid & 3;                       // 0..3
    const int hiHalf  = (lane & 16) ? 8 : 0;
    const int m = rowTile * 16 + (lane & 15);
    const int n = colTile * 16 + (lane & 15);

    const float* xrow = x + (size_t)m * F_IN;
    v8f acc = {};

    // ---- 15 full, unguarded k-steps: k in [0, 480) ----
    for (int kb = 0; kb < 480; kb += 32) {
        const int kA = kb + hiHalf;
        const float4 r0 = *(const float4*)(xrow + kA);
        const float4 r1 = *(const float4*)(xrow + kA + 4);
        const float4 r2 = *(const float4*)(xrow + kA + 16);
        const float4 r3 = *(const float4*)(xrow + kA + 20);

        v16bf a;
        a[0]  = (__bf16)r0.x; a[1]  = (__bf16)r0.y; a[2]  = (__bf16)r0.z; a[3]  = (__bf16)r0.w;
        a[4]  = (__bf16)r1.x; a[5]  = (__bf16)r1.y; a[6]  = (__bf16)r1.z; a[7]  = (__bf16)r1.w;
        a[8]  = (__bf16)r2.x; a[9]  = (__bf16)r2.y; a[10] = (__bf16)r2.z; a[11] = (__bf16)r2.w;
        a[12] = (__bf16)r3.x; a[13] = (__bf16)r3.y; a[14] = (__bf16)r3.z; a[15] = (__bf16)r3.w;

        v16bf b;
#pragma unroll
        for (int j = 0; j < 16; ++j) {
            int k = kb + j + (j & 8) + hiHalf;         // <= 479, in bounds
            b[j] = (__bf16)W1[k * H_DIM + n];
        }
        acc = __builtin_amdgcn_wmma_f32_16x16x32_bf16(
            false, a, false, b, (short)0, acc, false, false);
    }

    // ---- peeled tail: kb = 480, valid k in [480, 500) ----
    {
        const int kA = 480 + hiHalf;
        const float4 r0 = *(const float4*)(xrow + kA);       // 480..487 / 488..495
        const float4 r1 = *(const float4*)(xrow + kA + 4);
        const float4 r2 = *(const float4*)(xrow + 496);      // clamped last float4
        const float fa2[4] = {r2.x, r2.y, r2.z, r2.w};

        v16bf a;
        a[0] = (__bf16)r0.x; a[1] = (__bf16)r0.y; a[2] = (__bf16)r0.z; a[3] = (__bf16)r0.w;
        a[4] = (__bf16)r1.x; a[5] = (__bf16)r1.y; a[6] = (__bf16)r1.z; a[7] = (__bf16)r1.w;
#pragma unroll
        for (int j = 0; j < 8; ++j) {
            int k1 = kA + 16 + j;                      // 496..503 or 504..511
            float val = (j < 4) ? fa2[j] : 0.0f;       // compile-time select
            float f1  = (k1 < F_IN) ? val : 0.0f;      // v_cndmask, no branch
            a[j + 8] = (__bf16)f1;
        }

        v16bf b;                                       // B garbage at k>=500 is
#pragma unroll                                         // multiplied by A zeros
        for (int j = 0; j < 16; ++j) {
            int k  = 480 + j + (j & 8) + hiHalf;
            int kc = (k < F_IN) ? k : (F_IN - 1);      // clamp: in-bounds load
            b[j] = (__bf16)W1[kc * H_DIM + n];
        }
        acc = __builtin_amdgcn_wmma_f32_16x16x32_bf16(
            false, a, false, b, (short)0, acc, false, false);
    }

    const float bias = b1[n];
#pragma unroll
    for (int r = 0; r < 8; ++r) {
        int mr = rowTile * 16 + r + hiHalf;
        float v = fmaxf(acc[r] + bias, 0.0f);
        h1[(size_t)mr * H_DIM + n] = (__bf16)v;
    }
}

// ---------------------------------------------------------------------------
// GEMM2: h[N x 40] = h1[N x 64](bf16) @ W2[64 x 40] + b2 ; also copied to h0.
// Col tiles padded to 48: columns 40..47 compute garbage from a clamped W2
// column and are simply never stored (no per-element masking needed).
// ---------------------------------------------------------------------------
__global__ __launch_bounds__(256) void appnp_gemm2(const __bf16* __restrict__ h1,
                                                   const float* __restrict__ W2,
                                                   const float* __restrict__ b2,
                                                   float* __restrict__ h,
                                                   float* __restrict__ h0) {
    const int lane = threadIdx.x & 31;
    const int wid  = threadIdx.x >> 5;
    const int tile = blockIdx.x * 8 + wid;
    const int nTiles = (N_NODES / 16) * 3;            // 18750
    if (tile >= nTiles) return;
    const int rowTile = tile / 3;
    const int colTile = tile % 3;
    const int hiHalf  = (lane & 16) ? 8 : 0;
    const int m = rowTile * 16 + (lane & 15);
    const int n = colTile * 16 + (lane & 15);
    const int ncl = (n < C_DIM) ? n : (C_DIM - 1);    // clamped column

    const __bf16* hrow = h1 + (size_t)m * H_DIM;
    v8f acc = {};
#pragma unroll
    for (int kb = 0; kb < H_DIM; kb += 32) {
        const int kA = kb + hiHalf;
        const v8bf lo = *(const v8bf*)(hrow + kA);        // 16B load
        const v8bf hi = *(const v8bf*)(hrow + kA + 16);   // 16B load
        v16bf a;
#pragma unroll
        for (int j = 0; j < 8; ++j) { a[j] = lo[j]; a[j + 8] = hi[j]; }

        v16bf b;
#pragma unroll
        for (int j = 0; j < 16; ++j) {
            int k = kb + j + (j & 8) + hiHalf;            // < 64, exact
            b[j] = (__bf16)W2[k * C_DIM + ncl];
        }
        acc = __builtin_amdgcn_wmma_f32_16x16x32_bf16(
            false, a, false, b, (short)0, acc, false, false);
    }

    if (n < C_DIM) {
        const float bias = b2[ncl];
#pragma unroll
        for (int r = 0; r < 8; ++r) {
            int mr = rowTile * 16 + r + hiHalf;
            float v = acc[r] + bias;
            h [(size_t)mr * C_DIM + n] = v;
            h0[(size_t)mr * C_DIM + n] = v;
        }
    }
}

// ---------------------------------------------------------------------------
// Degree / normalization
// ---------------------------------------------------------------------------
__global__ __launch_bounds__(256) void appnp_deg_init(float* __restrict__ deg) {
    int i = blockIdx.x * blockDim.x + threadIdx.x;
    if (i < N_NODES) deg[i] = 1.0f;                    // self loop
}

__global__ __launch_bounds__(256) void appnp_deg_count(const int* __restrict__ dst,
                                                       float* __restrict__ deg) {
    int e = blockIdx.x * blockDim.x + threadIdx.x;
    if (e < N_EDGES) unsafeAtomicAdd(&deg[dst[e]], 1.0f);
}

__global__ __launch_bounds__(256) void appnp_dinv(const float* __restrict__ deg,
                                                  float* __restrict__ dinv) {
    int i = blockIdx.x * blockDim.x + threadIdx.x;
    if (i < N_NODES) dinv[i] = rsqrtf(deg[i]);
}

// ---------------------------------------------------------------------------
// Propagation step part 1 (dense init, fully coalesced):
//   hn = ALPHA*h0 + (1-ALPHA)*dinv[i]^2 * h   (self-loop folded in)
// ---------------------------------------------------------------------------
__global__ __launch_bounds__(256) void appnp_prop_init(const float* __restrict__ h,
                                                       const float* __restrict__ h0,
                                                       const float* __restrict__ dinv,
                                                       float* __restrict__ hn) {
    int t = blockIdx.x * blockDim.x + threadIdx.x;     // N*40 threads, exact
    int i = t / C_DIM;
    float di = dinv[i];
    hn[t] = ALPHA * h0[t] + (1.0f - ALPHA) * di * di * h[t];
}

// ---------------------------------------------------------------------------
// Propagation step part 2: one edge per wave32, lanes cover the 40 channels.
//   hn[dst] += (1-ALPHA) * dinv[src]*dinv[dst] * h[src]
// ---------------------------------------------------------------------------
__global__ __launch_bounds__(256) void appnp_prop_edges(const int* __restrict__ src,
                                                        const int* __restrict__ dst,
                                                        const float* __restrict__ dinv,
                                                        const float* __restrict__ h,
                                                        float* __restrict__ hn) {
    const int lane = threadIdx.x & 31;
    const int e = blockIdx.x * 8 + (threadIdx.x >> 5);
    if (e >= N_EDGES) return;
    const int s = src[e];
    const int d = dst[e];
    const float w = (1.0f - ALPHA) * dinv[s] * dinv[d];
    const float* hs = h  + (size_t)s * C_DIM;
    float*       hd = hn + (size_t)d * C_DIM;
    unsafeAtomicAdd(&hd[lane], w * hs[lane]);          // lanes 0..31
    if (lane < C_DIM - 32)                             // lanes 0..7 -> ch 32..39
        unsafeAtomicAdd(&hd[lane + 32], w * hs[lane + 32]);
}

// ---------------------------------------------------------------------------
// log_softmax over 40 channels: one wave32 per row, shfl_xor reductions.
// ---------------------------------------------------------------------------
__global__ __launch_bounds__(256) void appnp_logsoftmax(const float* __restrict__ h,
                                                        float* __restrict__ out) {
    const int lane = threadIdx.x & 31;
    const int row = blockIdx.x * 8 + (threadIdx.x >> 5);
    if (row >= N_NODES) return;
    const float* r = h + (size_t)row * C_DIM;
    float v0 = r[lane];
    float v1 = (lane < C_DIM - 32) ? r[lane + 32] : -3.4e38f;

    float m = fmaxf(v0, v1);
#pragma unroll
    for (int off = 16; off > 0; off >>= 1) m = fmaxf(m, __shfl_xor(m, off, 32));

    float s = __expf(v0 - m) + ((lane < C_DIM - 32) ? __expf(v1 - m) : 0.0f);
#pragma unroll
    for (int off = 16; off > 0; off >>= 1) s += __shfl_xor(s, off, 32);

    float lse = m + __logf(s);
    float* o = out + (size_t)row * C_DIM;
    o[lane] = v0 - lse;
    if (lane < C_DIM - 32) o[lane + 32] = v1 - lse;
}

// ---------------------------------------------------------------------------
// Host-side orchestration
// ---------------------------------------------------------------------------
extern "C" void kernel_launch(void* const* d_in, const int* in_sizes, int n_in,
                              void* d_out, int out_size, void* d_ws, size_t ws_size,
                              hipStream_t stream) {
    const float* x  = (const float*)d_in[0];
    const float* W1 = (const float*)d_in[1];
    const float* b1 = (const float*)d_in[2];
    const float* W2 = (const float*)d_in[3];
    const float* b2 = (const float*)d_in[4];
    const int*   ei = (const int*)d_in[5];       // [2 x E] flat: src then dst
    const int*   src = ei;
    const int*   dst = ei + N_EDGES;
    float* out = (float*)d_out;

    // Workspace layout (bytes, 256-aligned)
    char* ws = (char*)d_ws;
    __bf16* h1  = (__bf16*)ws;                                  // N*64*2  = 12.8 MB
    float* hA   = (float*)(ws + 12800000);                      // N*40*4  = 16 MB
    float* hB   = (float*)(ws + 12800000 + 16000000);           // 16 MB
    float* h0   = (float*)(ws + 12800000 + 2 * 16000000);       // 16 MB
    float* deg  = (float*)(ws + 12800000 + 3 * 16000000);       // 400 KB
    float* dinv = (float*)(ws + 12800000 + 3 * 16000000 + 400128);

    // MLP
    appnp_gemm1<<<N_NODES / 32, 256, 0, stream>>>(x, W1, b1, h1);
    appnp_gemm2<<<((N_NODES / 16) * 3 + 7) / 8, 256, 0, stream>>>(h1, W2, b2, hA, h0);

    // gcn_norm with self loops
    appnp_deg_init<<<(N_NODES + 255) / 256, 256, 0, stream>>>(deg);
    appnp_deg_count<<<(N_EDGES + 255) / 256, 256, 0, stream>>>(dst, deg);
    appnp_dinv<<<(N_NODES + 255) / 256, 256, 0, stream>>>(deg, dinv);

    // K-step propagation (ping-pong; K even -> result ends in hA)
    float* cur = hA;
    float* nxt = hB;
    for (int k = 0; k < K_STEPS; ++k) {
        appnp_prop_init<<<(N_NODES * C_DIM) / 256, 256, 0, stream>>>(cur, h0, dinv, nxt);
        appnp_prop_edges<<<(N_EDGES + 7) / 8, 256, 0, stream>>>(src, dst, dinv, cur, nxt);
        float* t = cur; cur = nxt; nxt = t;
    }

    appnp_logsoftmax<<<(N_NODES + 7) / 8, 256, 0, stream>>>(cur, out);
}
